// AxisAttention_38817914421346
// MI455X (gfx1250) — matrix operand order repaired and checked by reading, not compiled, not run on hardware
//
#include <hip/hip_runtime.h>

// ---------------------------------------------------------------------------
// AxisAttention for MI455X (gfx1250), wave32 + WMMA + TDM.
//   B=4, N=2048, D(=DQ=DKV=DA)=512.  ~51.6 GFLOP, ~55MB unique data (fits in
//   192MB L2) -> compute bound; all GEMMs routed through
//   v_wmma_f32_16x16x32_f16; q tile staged to LDS via tensor_load_to_lds.
// ---------------------------------------------------------------------------

typedef __attribute__((ext_vector_type(16))) _Float16 v16h;
typedef __attribute__((ext_vector_type(8)))  _Float16 v8h;
typedef __attribute__((ext_vector_type(8)))  float    v8f;
typedef __attribute__((ext_vector_type(4)))  unsigned int u32x4;
typedef __attribute__((ext_vector_type(8)))  int      i32x8;
typedef __attribute__((ext_vector_type(4)))  int      i32x4;

#define DEV static __device__ __forceinline__

DEV v16h cat8(v8h lo, v8h hi) {
  return __builtin_shufflevector(lo, hi, 0, 1, 2, 3, 4, 5, 6, 7,
                                         8, 9, 10, 11, 12, 13, 14, 15);
}

// A fragment (16x32 f16): rows row0..row0+15, K = k0..k0+31, row-major src.
// ISA layout: lanes 0-15 hold M=lane, K 0..7 & 16..23; lanes 16-31 K 8..15 & 24..31.
DEV v16h load_a_frag(const _Float16* base, int row0, int k0, int ld) {
  const int lane = threadIdx.x & 31;
  const _Float16* p = base + (size_t)(row0 + (lane & 15)) * ld + k0 + ((lane >> 4) << 3);
  v8h lo = *(const v8h*)p;        // K = ksub .. ksub+7
  v8h hi = *(const v8h*)(p + 16); // K = 16+ksub .. 16+ksub+7
  return cat8(lo, hi);
}

// B fragment (32x16 f16): B[k][n] with source S[n][k] row-major (K contiguous).
// ISA layout: lanes 0-15 hold N=lane, K 0..7 & 16..23; lanes 16-31 K 8..15 & 24..31.
DEV v16h load_b_frag(const _Float16* base, int n0, int k0, int ld) {
  const int lane = threadIdx.x & 31;
  const _Float16* p = base + (size_t)(n0 + (lane & 15)) * ld + k0 + ((lane >> 4) << 3);
  v8h lo = *(const v8h*)p;
  v8h hi = *(const v8h*)(p + 16);
  return cat8(lo, hi);
}

DEV v8f wmma_f16(v16h a, v16h b, v8f c) {
  // v_wmma_f32_16x16x32_f16: (neg_a, A, neg_b, B, c_mod, C, reuse_a, reuse_b)
  return __builtin_amdgcn_wmma_f32_16x16x32_f16(false, a, false, b,
                                                (short)0, c, false, false);
}

// Tensor Data Mover: 2D tile load (rows x cols of 2-byte elements, row-major,
// row stride = ld elements) from global -> LDS.  D# per ISA ch.8.3/8.4.
DEV void tdm_load_2d_f16(unsigned lds_off, const _Float16* gsrc,
                         unsigned rows, unsigned cols, unsigned ld) {
  const unsigned long long ga = (unsigned long long)(size_t)gsrc;
  u32x4 g0;
  g0[0] = 1u;                                      // count=1, user descriptor
  g0[1] = lds_off;                                 // LDS byte address
  g0[2] = (unsigned)(ga & 0xFFFFFFFFu);            // global_addr[31:0]
  g0[3] = (unsigned)((ga >> 32) & 0x01FFFFFFu)     // global_addr[56:32]
        | (2u << 30);                              // type = 2 ("image")
  i32x8 g1;
  g1[0] = (int)(1u << 16);                         // wg_mask=0, data_size=1 (2B)
  g1[1] = (int)(cols << 16);                       // tensor_dim0[15:0]
  g1[2] = (int)((cols >> 16) | (8192u << 16));     // dim0 hi | tensor_dim1 lo
  g1[3] = (int)(cols << 16);                       // dim1 hi=0 | tile_dim0
  g1[4] = (int)rows;                               // tile_dim1 | tile_dim2=0
  g1[5] = (int)ld;                                 // tensor_dim0_stride[31:0]
  g1[6] = 0;                                       // stride hi | dim1_stride lo
  g1[7] = 0;
  i32x4 gz = {0, 0, 0, 0};                         // 2D: groups 2/3 unused
#if defined(__clang_major__) && __clang_major__ >= 23
  i32x8 gz8 = {0, 0, 0, 0, 0, 0, 0, 0};
  __builtin_amdgcn_tensor_load_to_lds(g0, g1, gz, gz, gz8, 0);
#else
  __builtin_amdgcn_tensor_load_to_lds(g0, g1, gz, gz, 0);
#endif
}

// ---------------------------------------------------------------------------
// Kernel 1: projections.  z=0: q=query@Wq+bq  z=1: k=kv@Wk+bk  z=2: v (stored
// transposed as vT[b][d][n] so attn@v B-fragments are contiguous loads).
// Tile: 128(M) x 64(N), 8 waves, wave = 16 rows x 64 cols = 4 accumulators.
// ---------------------------------------------------------------------------
__global__ __launch_bounds__(256)
void proj_kernel(const float* __restrict__ query, const float* __restrict__ kv,
                 const float* __restrict__ Wq, const float* __restrict__ bq,
                 const float* __restrict__ Wk, const float* __restrict__ bk,
                 const float* __restrict__ Wv, const float* __restrict__ bv,
                 _Float16* __restrict__ q16, _Float16* __restrict__ k16,
                 _Float16* __restrict__ vT16) {
  constexpr int D = 512, NTOK = 2048;
  const int z = blockIdx.z;
  const float* X    = (z == 0) ? query : kv;
  const float* W    = (z == 0) ? Wq : (z == 1) ? Wk : Wv;
  const float* bias = (z == 0) ? bq : (z == 1) ? bk : bv;

  __shared__ __align__(16) _Float16 ldsA[128][32];   // X tile (f32 -> f16)
  __shared__ __align__(16) _Float16 ldsBt[64][32];   // W tile, transposed [n][k]

  const int t = threadIdx.x, wave = t >> 5, lane = t & 31;
  const int row0 = blockIdx.x * 128;
  const int col0 = blockIdx.y * 64;
  const int wrow = row0 + wave * 16;

  v8f acc[4] = {};
  for (int kk = 0; kk < D; kk += 32) {
    __syncthreads();
    for (int i = t; i < 128 * 32; i += 256) {
      int r = i >> 5, c = i & 31;
      ldsA[r][c] = (_Float16)X[(size_t)(row0 + r) * D + kk + c];
    }
    for (int i = t; i < 32 * 64; i += 256) {
      int k = i >> 6, n = i & 63;                       // coalesced along n
      ldsBt[n][k] = (_Float16)W[(size_t)(kk + k) * D + col0 + n];
    }
    __syncthreads();
    v16h a = load_a_frag(&ldsA[0][0], wave * 16, 0, 32);
#pragma unroll
    for (int j = 0; j < 4; ++j) {
      v16h b = load_b_frag(&ldsBt[0][0], j * 16, 0, 32);
      acc[j] = wmma_f16(a, b, acc[j]);
    }
  }

  const int hi = lane >> 4, nsub = lane & 15;
  const int mb = wrow + hi * 8;          // acc[j][i] is row mb+i, col col0+j*16+nsub
  const int batch = mb >> 11;
  const int nb = mb & (NTOK - 1);
#pragma unroll
  for (int j = 0; j < 4; ++j) {
    const int col = col0 + j * 16 + nsub;
    const float bb = bias[col];
    if (z == 2) {                         // transposed store: 8 contiguous halfs
      v8h vec;
#pragma unroll
      for (int i = 0; i < 8; ++i) vec[i] = (_Float16)(acc[j][i] + bb);
      *(v8h*)(vT16 + ((size_t)batch * D + col) * NTOK + nb) = vec;
    } else {
      _Float16* dst = (z == 0) ? q16 : k16;
#pragma unroll
      for (int i = 0; i < 8; ++i)
        dst[(size_t)(mb + i) * D + col] = (_Float16)(acc[j][i] + bb);
    }
  }
}

// ---------------------------------------------------------------------------
// Kernel 2: flash attention.  Block = 32 query rows of one batch, 8 waves.
// q tile (32x512 f16, 32KB) staged once via the Tensor Data Mover.  Per
// 128-key tile: wave w computes S[:, w*16:(w+1)*16] (full K=512 via WMMA; k
// rows ARE contiguous B-fragments since B = k^T), online softmax in LDS,
// then each wave owns a 64-wide D-slice of O and accumulates P@V from vT.
// ---------------------------------------------------------------------------
__global__ __launch_bounds__(256)
void attn_kernel(const _Float16* __restrict__ q16, const _Float16* __restrict__ k16,
                 const _Float16* __restrict__ vT16, _Float16* __restrict__ o16) {
  constexpr int D = 512, NTOK = 2048, BC = 128, BR = 32;
  constexpr float SCALE = 22.62741699796952f;   // sqrt(512): multiply, per ref

  const int t = threadIdx.x, wave = t >> 5, lane = t & 31;
  const int b = blockIdx.y;
  const int q0 = blockIdx.x * BR;
  const _Float16* qb = q16 + (size_t)b * NTOK * D;
  const _Float16* kb = k16 + (size_t)b * NTOK * D;
  const _Float16* vb = vT16 + (size_t)b * D * NTOK;   // vT[d][n]

  __shared__ __align__(16) _Float16 ldsQ[BR][D];      // 32 KB, TDM destination
  __shared__ __align__(16) float    Smat[BR][BC];     // 16 KB
  __shared__ __align__(16) _Float16 Pmat[BR][BC];     //  8 KB
  __shared__ float mrow[BR], lrow[BR], srow[BR];

  if (t < BR) { mrow[t] = -1e30f; lrow[t] = 0.f; }

  // --- TDM: stage this block's q tile into LDS (wave 0 issues & waits) -----
  if (wave == 0) {
    tdm_load_2d_f16((unsigned)(size_t)&ldsQ[0][0], qb + (size_t)q0 * D,
                    BR, D, D);
    __builtin_amdgcn_s_wait_tensorcnt(0);
  }
  __syncthreads();

  const int hi = lane >> 4, nsub = lane & 15;
  v8f O[2][4] = {};   // [row-tile][d-col-tile], wave's D slice = [wave*64, +64)

  for (int kt = 0; kt < NTOK; kt += BC) {
    // prefetch next key tile (one byte per row -> whole cacheline pulled)
    if (kt + BC < NTOK)
      __builtin_prefetch(kb + (size_t)(kt + BC + wave * 16 + nsub) * D, 0, 1);

    // --- scores: S = (q @ k^T) for this wave's 16 key columns -------------
    v8f S[2] = {};
#pragma unroll 1
    for (int kk = 0; kk < D; kk += 32) {
      v16h bf = load_b_frag(kb, kt + wave * 16, kk, D);
#pragma unroll
      for (int rt = 0; rt < 2; ++rt) {
        v16h a = load_a_frag(&ldsQ[0][0], rt * 16, kk, D);
        S[rt] = wmma_f16(a, bf, S[rt]);
      }
    }
    __syncthreads();   // previous tile's Pmat readers are done
#pragma unroll
    for (int rt = 0; rt < 2; ++rt)
#pragma unroll
      for (int i = 0; i < 8; ++i)
        Smat[rt * 16 + i + 8 * hi][wave * 16 + nsub] = S[rt][i] * SCALE;
    __syncthreads();

    // --- online softmax: one thread per query row -------------------------
    if (t < BR) {
      float m0 = mrow[t], mn = m0;
      for (int c = 0; c < BC; ++c) mn = fmaxf(mn, Smat[t][c]);
      float sc = __expf(m0 - mn);
      float ls = 0.f;
      for (int c = 0; c < BC; ++c) {
        float p = __expf(Smat[t][c] - mn);
        Pmat[t][c] = (_Float16)p;
        ls += p;
      }
      mrow[t] = mn;
      lrow[t] = lrow[t] * sc + ls;
      srow[t] = sc;
    }
    __syncthreads();

    // --- rescale O, accumulate O += P @ V slice ---------------------------
#pragma unroll
    for (int rt = 0; rt < 2; ++rt) {
      float f[8];
#pragma unroll
      for (int i = 0; i < 8; ++i) f[i] = srow[rt * 16 + i + 8 * hi];
#pragma unroll
      for (int ct = 0; ct < 4; ++ct)
#pragma unroll
        for (int i = 0; i < 8; ++i) O[rt][ct][i] *= f[i];
    }
#pragma unroll 1
    for (int kk = 0; kk < BC; kk += 32) {
      v16h bf[4];
#pragma unroll
      for (int ct = 0; ct < 4; ++ct)      // contiguous: vT rows are d-columns
        bf[ct] = load_b_frag(vb, wave * 64 + ct * 16, kt + kk, NTOK);
#pragma unroll
      for (int rt = 0; rt < 2; ++rt) {
        v16h a = load_a_frag(&Pmat[0][0], rt * 16, kk, BC);
#pragma unroll
        for (int ct = 0; ct < 4; ++ct)
          O[rt][ct] = wmma_f16(a, bf[ct], O[rt][ct]);
      }
    }
  }

  // --- normalize by l, store f16 ------------------------------------------
#pragma unroll
  for (int rt = 0; rt < 2; ++rt) {
    float f[8];
#pragma unroll
    for (int i = 0; i < 8; ++i) f[i] = 1.0f / lrow[rt * 16 + i + 8 * hi];
#pragma unroll
    for (int ct = 0; ct < 4; ++ct) {
      const int col = wave * 64 + ct * 16 + nsub;
#pragma unroll
      for (int i = 0; i < 8; ++i) {
        const int row = q0 + rt * 16 + i + 8 * hi;
        o16[((size_t)b * NTOK + row) * D + col] = (_Float16)(O[rt][ct][i] * f[i]);
      }
    }
  }
}

// ---------------------------------------------------------------------------
// Kernel 3: result = query + O @ Wo + bo  (f32 out).  O is already f16 and
// row-major, so A-fragments load straight from global; Wo staged transposed.
// ---------------------------------------------------------------------------
__global__ __launch_bounds__(256)
void outproj_kernel(const _Float16* __restrict__ o16, const float* __restrict__ Wo,
                    const float* __restrict__ bo, const float* __restrict__ query,
                    float* __restrict__ out) {
  constexpr int D = 512;
  __shared__ __align__(16) _Float16 ldsBt[64][32];
  const int t = threadIdx.x, wave = t >> 5, lane = t & 31;
  const int row0 = blockIdx.x * 128, col0 = blockIdx.y * 64;
  const int wrow = row0 + wave * 16;

  v8f acc[4] = {};
  for (int kk = 0; kk < D; kk += 32) {
    __syncthreads();
    for (int i = t; i < 32 * 64; i += 256) {
      int k = i >> 6, n = i & 63;
      ldsBt[n][k] = (_Float16)Wo[(size_t)(kk + k) * D + col0 + n];
    }
    __syncthreads();
    v16h a = load_a_frag(o16, wrow, kk, D);
#pragma unroll
    for (int j = 0; j < 4; ++j) {
      v16h b = load_b_frag(&ldsBt[0][0], j * 16, 0, 32);
      acc[j] = wmma_f16(a, b, acc[j]);
    }
  }

  const int hi = lane >> 4, nsub = lane & 15;
  const int mb = wrow + 8 * hi;
#pragma unroll
  for (int j = 0; j < 4; ++j) {
    const int col = col0 + j * 16 + nsub;
    const float bb = bo[col];
#pragma unroll
    for (int i = 0; i < 8; ++i) {
      const size_t idx = (size_t)(mb + i) * D + col;
      out[idx] = query[idx] + acc[j][i] + bb;
    }
  }
}

// ---------------------------------------------------------------------------
extern "C" void kernel_launch(void* const* d_in, const int* in_sizes, int n_in,
                              void* d_out, int out_size, void* d_ws, size_t ws_size,
                              hipStream_t stream) {
  (void)in_sizes; (void)n_in; (void)out_size; (void)ws_size;
  const float* query = (const float*)d_in[0];
  const float* kv    = (const float*)d_in[1];
  const float* Wq    = (const float*)d_in[2];
  const float* bq    = (const float*)d_in[3];
  const float* Wk    = (const float*)d_in[4];
  const float* bk    = (const float*)d_in[5];
  const float* Wv    = (const float*)d_in[6];
  const float* bv    = (const float*)d_in[7];
  const float* Wo    = (const float*)d_in[8];
  const float* bo    = (const float*)d_in[9];

  const size_t tokD = (size_t)4 * 2048 * 512;   // elements per (B,N,D) tensor
  _Float16* ws   = (_Float16*)d_ws;             // 4 * tokD * 2B = 33.5 MB
  _Float16* q16  = ws;
  _Float16* k16  = ws + tokD;
  _Float16* vT16 = ws + 2 * tokD;
  _Float16* o16  = ws + 3 * tokD;

  dim3 gProj(8192 / 128, 512 / 64, 3);          // (64, 8, 3)
  proj_kernel<<<gProj, 256, 0, stream>>>(query, kv, Wq, bq, Wk, bk, Wv, bv,
                                         q16, k16, vT16);

  dim3 gAttn(2048 / 32, 4);                     // (64, 4)
  attn_kernel<<<gAttn, 256, 0, stream>>>(q16, k16, vT16, o16);

  dim3 gOut(8192 / 128, 512 / 64);              // (64, 8)
  outproj_kernel<<<gOut, 256, 0, stream>>>(o16, Wo, bo, query, (float*)d_out);
}
